// xLSTM_12438225289800
// MI455X (gfx1250) — matrix operand order repaired and checked by reading, not compiled, not run on hardware
//
#include <hip/hip_runtime.h>
#include <math.h>

// ---------------------------------------------------------------------------
// Types for CDNA5 WMMA (wave32): v_wmma_f32_16x16x32_bf16
// ---------------------------------------------------------------------------
typedef __bf16 bf16;
typedef __attribute__((ext_vector_type(4)))  __bf16 v4bf;
typedef __attribute__((ext_vector_type(8)))  __bf16 v8bf;
typedef __attribute__((ext_vector_type(16))) __bf16 v16bf;
typedef __attribute__((ext_vector_type(8)))  float  v8f;

#define SHUF16(lo, hi) __builtin_shufflevector((lo), (hi), \
    0,1,2,3,4,5,6,7,8,9,10,11,12,13,14,15)

__device__ __forceinline__ v8f wmma_bf16(v16bf a, v16bf b, v8f c) {
  // (neg_a, A, neg_b, B, c_mod, C, reuse_a, reuse_b)
  return __builtin_amdgcn_wmma_f32_16x16x32_bf16(false, a, false, b,
                                                 (short)0, c, false, false);
}

// A-fragment (16x32 bf16, MxK): lane<16 -> M=lane, K=kb..kb+15 contiguous
//                               lane>=16 -> M=lane-16, K=kb+16..kb+31
__device__ __forceinline__ v16bf load_a_frag(const bf16* __restrict__ rowK8) {
  // rowK8 already points at  A + M*K + (half?16:0)
  v8bf lo = *(const v8bf*)(rowK8);
  v8bf hi = *(const v8bf*)(rowK8 + 8);
  return SHUF16(lo, hi);
}

// B-fragment (32x16 bf16, KxN, sourced from W[N][K] row-major so B[k][n]=W[n][k]):
// lane<16 -> N=lane, K=kb..kb+15 contiguous; lane>=16 -> N=lane-16, K=kb+16..kb+31
__device__ __forceinline__ v16bf load_b_frag(const bf16* __restrict__ rowK16) {
  // rowK16 already points at  W + N*K + (half?16:0)
  return *(const v16bf*)(rowK16);
}

// ---------------------------------------------------------------------------
// fp32 -> bf16 conversion
// ---------------------------------------------------------------------------
__global__ void k_cvt_bf16(const float* __restrict__ s, bf16* __restrict__ d, int n) {
  int i = blockIdx.x * blockDim.x + threadIdx.x;
  if (i < n) d[i] = (bf16)s[i];
}

// ---------------------------------------------------------------------------
// WMMA GEMM, 2x2 register blocked: each wave computes a 32x32 C block.
//   C[M,N] = A[M,K](bf16) * W[N,K](bf16)^T + bias[N]
// Per K-step: 2 A frags + 2 B frags -> 4 WMMAs (2 b128 loads per WMMA).
// K multiple of 32; M,N multiples of 32.
// ---------------------------------------------------------------------------
__global__ __launch_bounds__(256)
void k_gemm_bf16(const bf16* __restrict__ A, const bf16* __restrict__ W,
                 const float* __restrict__ bias, float* __restrict__ C,
                 int M, int N, int K) {
  int wave = (blockIdx.x * blockDim.x + threadIdx.x) >> 5;
  int lane = threadIdx.x & 31;
  int ntn = N >> 5;                       // 32-wide N strips
  int ntiles = (M >> 5) * ntn;
  if (wave >= ntiles) return;             // wave-uniform: EXEC stays all-ones
  int m0 = (wave / ntn) << 5;
  int n0 = (wave % ntn) << 5;
  int half = lane >> 4;
  int l = lane & 15;

  const bf16* arow0 = A + (size_t)(m0 + l) * K + half * 16;
  const bf16* arow1 = arow0 + (size_t)16 * K;
  const bf16* brow0 = W + (size_t)(n0 + l) * K + half * 16;
  const bf16* brow1 = brow0 + (size_t)16 * K;

  v8f acc00 = {}, acc01 = {}, acc10 = {}, acc11 = {};
  for (int k = 0; k < K; k += 32) {
    v16bf a0 = load_a_frag(arow0 + k);
    v16bf a1 = load_a_frag(arow1 + k);
    v16bf b0 = load_b_frag(brow0 + k);
    v16bf b1 = load_b_frag(brow1 + k);
    acc00 = wmma_bf16(a0, b0, acc00);
    acc01 = wmma_bf16(a0, b1, acc01);
    acc10 = wmma_bf16(a1, b0, acc10);
    acc11 = wmma_bf16(a1, b1, acc11);
  }
  float bv0 = bias ? bias[n0 + l] : 0.0f;
  float bv1 = bias ? bias[n0 + 16 + l] : 0.0f;
  int rbase = half * 8;                   // lanes>=16 hold rows M=8..15 of a tile
#pragma unroll
  for (int r = 0; r < 8; ++r) {
    size_t row0 = (size_t)(m0 + rbase + r) * N;
    size_t row1 = (size_t)(m0 + 16 + rbase + r) * N;
    C[row0 + n0 + l]      = acc00[r] + bv0;
    C[row0 + n0 + 16 + l] = acc01[r] + bv1;
    C[row1 + n0 + l]      = acc10[r] + bv0;
    C[row1 + n0 + 16 + l] = acc11[r] + bv1;
  }
}

// ---------------------------------------------------------------------------
// Persistent sLSTM scan.
//  pre    : [B*T, 4H] fp32   (x @ wih^T + b, all timesteps)
//  whh    : [4H, H]   bf16
//  hstate : [16, H]   bf16   (rows 8..15 stay zero = M padding)
//  hs     : [B*T, H]  bf16   output hidden sequence
// 16 workgroups x 128 threads; WG wg owns hidden columns [wg*64, wg*64+64).
// Wave g computes gate g's four 16-wide subtiles, sharing one A fragment
// (the h state) across 4 WMMAs per K-step. Cell state c lives in registers
// (stable thread->element mapping). Device-wide phase barrier per timestep.
// ---------------------------------------------------------------------------
#define SCAN_NWG 16

__global__ __launch_bounds__(128)
void k_scan(const float* __restrict__ pre, const bf16* __restrict__ whh,
            bf16* __restrict__ hstate, bf16* __restrict__ hs,
            int* __restrict__ bar, int T) {
  const int H = 1024;
  const int wg = blockIdx.x;
  const int tid = threadIdx.x;
  const int lane = tid & 31;
  const int wave = tid >> 5;              // == gate index for this wave
  const int half = lane >> 4;
  const int l = lane & 15;
  const int colwg = wg * 64;

  __shared__ float gbuf[4][8][64];        // [gate][batch][col-in-chunk], 8 KB

  // B rows for the 4 subtiles (col offsets 0,16,32,48) of gate `wave`
  const bf16* brow0 = whh + (size_t)(wave * H + colwg +  0 + l) * H + half * 16;
  const bf16* brow1 = whh + (size_t)(wave * H + colwg + 16 + l) * H + half * 16;
  const bf16* brow2 = whh + (size_t)(wave * H + colwg + 32 + l) * H + half * 16;
  const bf16* brow3 = whh + (size_t)(wave * H + colwg + 48 + l) * H + half * 16;
  const bf16* arow  = hstate + (size_t)l * H + half * 16;

  // Activation mapping: batch = tid>>4, columns = (tid&15)*4 .. +3 (coalesced)
  const int ab = tid >> 4;
  const int ac = (tid & 15) * 4;
  const int agc = colwg + ac;
  float cc[4] = {0.0f, 0.0f, 0.0f, 0.0f};

  for (int t = 0; t < T; ++t) {
    v8f acc0 = {}, acc1 = {}, acc2 = {}, acc3 = {};
    for (int k = 0; k < H; k += 32) {
      v16bf av = load_a_frag(arow + k);
      acc0 = wmma_bf16(av, load_b_frag(brow0 + k), acc0);
      acc1 = wmma_bf16(av, load_b_frag(brow1 + k), acc1);
      acc2 = wmma_bf16(av, load_b_frag(brow2 + k), acc2);
      acc3 = wmma_bf16(av, load_b_frag(brow3 + k), acc3);
    }
    if (half == 0) {                      // lanes 0..15 hold batch rows 0..7
#pragma unroll
      for (int r = 0; r < 8; ++r) {
        gbuf[wave][r][ 0 + l] = acc0[r];
        gbuf[wave][r][16 + l] = acc1[r];
        gbuf[wave][r][32 + l] = acc2[r];
        gbuf[wave][r][48 + l] = acc3[r];
      }
    }
    __syncthreads();

    {
      const float* prow = pre + (size_t)(ab * T + t) * (4 * H);
      v4bf hv4;
#pragma unroll
      for (int j = 0; j < 4; ++j) {
        float vi = gbuf[0][ab][ac + j] + prow[0 * H + agc + j];
        float vf = gbuf[1][ab][ac + j] + prow[1 * H + agc + j];
        float vg = gbuf[2][ab][ac + j] + prow[2 * H + agc + j];
        float vo = gbuf[3][ab][ac + j] + prow[3 * H + agc + j];
        cc[j] = expf(vf) * cc[j] + expf(vi) * tanhf(vg);
        float hv = (1.0f / (1.0f + expf(-vo))) * tanhf(cc[j]);
        hv4[j] = (bf16)hv;
      }
      *(v4bf*)(hstate + (size_t)ab * H + agc) = hv4;            // 8B store
      *(v4bf*)(hs + (size_t)(ab * T + t) * H + agc) = hv4;      // 8B store
    }
    __syncthreads();
    __threadfence();                      // release hstate writes device-wide

    if (tid == 0) {                       // phase barrier (monotonic counter)
      __hip_atomic_fetch_add(bar, 1, __ATOMIC_ACQ_REL, __HIP_MEMORY_SCOPE_AGENT);
      const int target = (t + 1) * (int)gridDim.x;
      while (__hip_atomic_load(bar, __ATOMIC_ACQUIRE, __HIP_MEMORY_SCOPE_AGENT) < target) {
        __builtin_amdgcn_s_sleep(2);
      }
    }
    __syncthreads();
    __threadfence();                      // acquire other WGs' hstate writes
  }
}

// ---------------------------------------------------------------------------
// GELU(exact erf) + residual + LayerNorm; row per (b,t); D = 512, 128 threads.
// seq (fp32) updated in place; seq_bf written for the next bf16 GEMM.
// ---------------------------------------------------------------------------
__global__ __launch_bounds__(128)
void k_gelu_res_ln(const float* __restrict__ gemm, const float* __restrict__ pb,
                   const float* __restrict__ lg, const float* __restrict__ lb,
                   float* __restrict__ seq, bf16* __restrict__ seq_bf, int D) {
  const int row = blockIdx.x;
  const int tid = threadIdx.x;
  __shared__ float red[128];
  const float* grow = gemm + (size_t)row * D;
  float* srow = seq + (size_t)row * D;
  bf16* brow = seq_bf + (size_t)row * D;

  float vals[4];
  float s = 0.0f;
#pragma unroll
  for (int i = 0; i < 4; ++i) {
    int c = tid + i * 128;
    float xv = grow[c] + pb[c];
    float gel = 0.5f * xv * (1.0f + erff(xv * 0.70710678118654752f));
    float v = gel + srow[c];
    vals[i] = v;
    s += v;
  }
  red[tid] = s; __syncthreads();
  for (int off = 64; off > 0; off >>= 1) {
    if (tid < off) red[tid] += red[tid + off];
    __syncthreads();
  }
  float mean = red[0] * (1.0f / 512.0f);
  __syncthreads();
  float s2 = 0.0f;
#pragma unroll
  for (int i = 0; i < 4; ++i) { float d = vals[i] - mean; s2 += d * d; }
  red[tid] = s2; __syncthreads();
  for (int off = 64; off > 0; off >>= 1) {
    if (tid < off) red[tid] += red[tid + off];
    __syncthreads();
  }
  float rstd = rsqrtf(red[0] * (1.0f / 512.0f) + 1e-5f);
#pragma unroll
  for (int i = 0; i < 4; ++i) {
    int c = tid + i * 128;
    float y = (vals[i] - mean) * rstd * lg[c] + lb[c];
    srow[c] = y;
    brow[c] = (bf16)y;
  }
}

// ---------------------------------------------------------------------------
// Host orchestration
// ---------------------------------------------------------------------------
extern "C" void kernel_launch(void* const* d_in, const int* in_sizes, int n_in,
                              void* d_out, int out_size, void* d_ws, size_t ws_size,
                              hipStream_t stream) {
  (void)in_sizes; (void)n_in; (void)out_size; (void)ws_size;
  const float* x     = (const float*)d_in[0];
  const float* wih0  = (const float*)d_in[1];
  const float* whh0  = (const float*)d_in[2];
  const float* b0    = (const float*)d_in[3];
  const float* wih1  = (const float*)d_in[4];
  const float* whh1  = (const float*)d_in[5];
  const float* b1    = (const float*)d_in[6];
  const float* projw = (const float*)d_in[7];
  const float* projb = (const float*)d_in[8];
  const float* lng   = (const float*)d_in[9];
  const float* lnb   = (const float*)d_in[10];
  const float* outw  = (const float*)d_in[11];
  const float* outb  = (const float*)d_in[12];

  const int B = 8, T = 256, D = 512, H = 1024, O = 256, NB = 2;
  const int M = B * T;  // 2048 rows

  char* ws = (char*)d_ws;
  float* pre    = (float*)(ws + 0);          //  2048*4096 f32 = 32 MB
  bf16*  w_in   = (bf16*)(ws + 33554432);    //  up to 4096*1024 bf16 = 8 MB
  bf16*  w_rec  = (bf16*)(ws + 41943040);    //  4096*1024 bf16 = 8 MB
  float* seq_f  = (float*)(ws + 50331648);   //  2048*512 f32 = 4 MB
  bf16*  seq_b  = (bf16*)(ws + 54525952);    //  2048*512 bf16 = 2 MB
  bf16*  h1     = (bf16*)(ws + 56623104);    //  2048*1024 bf16 = 4 MB
  bf16*  h2     = (bf16*)(ws + 60817408);    //  2048*1024 bf16 = 4 MB
  float* projt  = (float*)(ws + 65011712);   //  2048*512 f32 = 4 MB
  bf16*  hstate = (bf16*)(ws + 69206016);    //  16*1024 bf16 = 32 KB
  int*   bar    = (int*)(ws + 69238784);     //  4 B

  auto cvt = [&](const float* s, bf16* d, int n) {
    k_cvt_bf16<<<(n + 255) / 256, 256, 0, stream>>>(s, d, n);
  };
  auto gemm = [&](const bf16* A, const bf16* W, const float* bias, float* C,
                  int Mm, int Nn, int Kk) {
    int ntiles = (Mm / 32) * (Nn / 32);   // 2x2-blocked: one wave per 32x32
    k_gemm_bf16<<<(ntiles + 7) / 8, 256, 0, stream>>>(A, W, bias, C, Mm, Nn, Kk);
  };
  auto scan = [&](bf16* hs) {
    hipMemsetAsync(hstate, 0, (size_t)16 * H * sizeof(bf16), stream);
    hipMemsetAsync(bar, 0, sizeof(int), stream);
    k_scan<<<SCAN_NWG, 128, 0, stream>>>(pre, w_rec, hstate, hs, bar, T);
  };

  // seq = x (fp32 residual path) + bf16 copy for GEMM input
  hipMemcpyAsync(seq_f, x, (size_t)M * D * sizeof(float),
                 hipMemcpyDeviceToDevice, stream);
  cvt(x, seq_b, M * D);

  for (int blk = 0; blk < NB; ++blk) {
    // ---- sLSTM layer 0 (D -> H) ----
    cvt(wih0 + (size_t)blk * 4 * H * D, w_in, 4 * H * D);
    gemm(seq_b, w_in, b0 + blk * 4 * H, pre, M, 4 * H, D);
    cvt(whh0 + (size_t)blk * 4 * H * H, w_rec, 4 * H * H);
    scan(h1);
    // ---- sLSTM layer 1 (H -> H) ----
    cvt(wih1 + (size_t)blk * 4 * H * H, w_in, 4 * H * H);
    gemm(h1, w_in, b1 + blk * 4 * H, pre, M, 4 * H, H);
    cvt(whh1 + (size_t)blk * 4 * H * H, w_rec, 4 * H * H);
    scan(h2);
    // ---- projection + GELU + residual + LayerNorm ----
    cvt(projw + (size_t)blk * D * H, w_in, D * H);
    gemm(h2, w_in, nullptr, projt, M, D, H);
    k_gelu_res_ln<<<M, 128, 0, stream>>>(projt, projb + blk * D,
                                         lng + blk * D, lnb + blk * D,
                                         seq_f, seq_b, D);
  }

  // ---- final output layer (D -> O), writes fp32 d_out [B,T,O] ----
  cvt(outw, w_in, O * D);
  gemm(seq_b, w_in, outb, (float*)d_out, M, O, D);
}
